// EdgeGATRegressor_66013647339603
// MI455X (gfx1250) — matrix operand order repaired and checked by reading, not compiled, not run on hardware
//
#include <hip/hip_runtime.h>

// ---------------- problem constants ----------------
constexpr int   NN   = 50000;          // nodes
constexpr int   NE   = 800000;         // edges (before self loops)
constexpr int   ET   = NE + NN;        // edges incl. self loops
constexpr int   FIN  = 128;            // input channels
constexpr int   H1   = 4,  C1 = 32;    // layer1 heads / per-head dim (H1*C1 = 128)
constexpr int   HC1  = H1 * C1;        // 128
constexpr int   H2   = 1,  C2 = 64;    // layer2
constexpr int   HC2  = H2 * C2;        // 64
constexpr int   KDIM = 128;            // GEMM K for both layers
constexpr int   STRIPS = NN / 16;      // 3125 exact
constexpr float NEG_SLOPE = 0.2f;
constexpr float EPSF = 1e-16f;

// ---------------- WMMA vector types ----------------
typedef __bf16 v16bf __attribute__((ext_vector_type(16)));
typedef __bf16 bf16x8 __attribute__((ext_vector_type(8)));
typedef float  v8f   __attribute__((ext_vector_type(8)));

// ---------------- small helpers ----------------
__device__ __forceinline__ unsigned short f2bf(float f) {
  unsigned u = __float_as_uint(f);
  unsigned r = ((u >> 16) & 1u) + 0x7FFFu;      // round to nearest even
  return (unsigned short)((u + r) >> 16);
}
__device__ __forceinline__ unsigned fenc(float f) {   // order-preserving f32->u32
  unsigned u = __float_as_uint(f);
  return (u & 0x80000000u) ? ~u : (u | 0x80000000u);
}
__device__ __forceinline__ float fdec(unsigned u) {
  return __uint_as_float((u & 0x80000000u) ? (u ^ 0x80000000u) : ~u);
}
constexpr unsigned ENC_NEG_INF = 0x007FFFFFu;         // fenc(-inf)

// ---------------- init kernels ----------------
__global__ void fill_u32(unsigned* __restrict__ p, unsigned v, long n) {
  long t = (long)blockIdx.x * blockDim.x + threadIdx.x;
  if (t < n) p[t] = v;
}

// ---------------- precision / layout prep ----------------
__global__ void cvt_bf16(const float* __restrict__ src, unsigned short* __restrict__ dst, long n) {
  long t = (long)blockIdx.x * blockDim.x + threadIdx.x;
  if (t < n) dst[t] = f2bf(src[t]);
}
// W[K][Nout] row-major f32 -> Wt[n][k] bf16 (column-contiguous in K for B fragments)
__global__ void cvt_transpose_w(const float* __restrict__ W, unsigned short* __restrict__ Wt,
                                int K, int Nout) {
  int t = blockIdx.x * blockDim.x + threadIdx.x;
  if (t >= K * Nout) return;
  int k = t / Nout, n = t - k * Nout;
  Wt[(long)n * K + k] = f2bf(W[(long)k * Nout + n]);
}

// ---------------- WMMA GEMM: H = A(bf16 [NN,128]) @ W, f32 out ----------------
template<int NOUT>
__global__ __launch_bounds__(128) void wmma_gemm_bf16(const unsigned short* __restrict__ Ab,
                                                      const unsigned short* __restrict__ Wt,
                                                      float* __restrict__ Hout) {
  constexpr int NT = NOUT / 16;
  const int wave = blockIdx.x * (blockDim.x >> 5) + (threadIdx.x >> 5);
  if (wave >= STRIPS) return;                 // wave-uniform: EXEC stays all-1 for WMMA
  const int  lane = threadIdx.x & 31;
  const int  m16  = lane & 15;
  const bool hi   = lane >= 16;
  const long row  = (long)wave * 16;

  v8f zero;
#pragma unroll
  for (int i = 0; i < 8; ++i) zero[i] = 0.0f;
  v8f acc[NT];
#pragma unroll
  for (int t = 0; t < NT; ++t) acc[t] = zero;

  const __bf16* A = (const __bf16*)Ab;
  const __bf16* B = (const __bf16*)Wt;

#pragma unroll
  for (int kb = 0; kb < KDIM; kb += 32) {
    // A fragment: lane m16 (lo half) holds K = kb+[0..15]; hi half K = kb+[16..31]
    union { v16bf v; bf16x8 h[2]; } af;
    const __bf16* ap = A + (row + m16) * KDIM + kb + (hi ? 16 : 0);
    af.h[0] = *(const bf16x8*)(ap);
    af.h[1] = *(const bf16x8*)(ap + 8);
#pragma unroll
    for (int t = 0; t < NT; ++t) {
      // B fragment: lane holds column (t*16+m16), 16 contiguous K (lo half kb..kb+15, hi half +16)
      const __bf16* bp = B + (long)(t * 16 + m16) * KDIM + kb + (hi ? 16 : 0);
      v16bf bf = *(const v16bf*)bp;
      acc[t] = __builtin_amdgcn_wmma_f32_16x16x32_bf16(
          false, af.v, false, bf, (short)0, acc[t], false, false);
    }
  }
  // D layout: VGPR r -> row r (lanes 0-15) / row 8+r (lanes 16-31), col = lane&15
  const int rbase = hi ? 8 : 0;
#pragma unroll
  for (int t = 0; t < NT; ++t)
#pragma unroll
    for (int r = 0; r < 8; ++r)
      Hout[(row + rbase + r) * NOUT + t * 16 + m16] = acc[t][r];
}

// ---------------- per-node attention scores ----------------
__global__ void node_scores(const float* __restrict__ Hf, const float* __restrict__ ws,
                            const float* __restrict__ wd, float* __restrict__ as_,
                            float* __restrict__ ad_, int Hh, int C) {
  int t = blockIdx.x * blockDim.x + threadIdx.x;
  if (t >= NN * Hh) return;
  int n = t / Hh, h = t - n * Hh;
  const float* hp = Hf + (long)n * Hh * C + (long)h * C;
  float s = 0.f, d = 0.f;
  for (int c = 0; c < C; ++c) { float v = hp[c]; s += v * ws[h * C + c]; d += v * wd[h * C + c]; }
  as_[t] = s; ad_[t] = d;
}

// ---------------- edge pass 1: leaky-relu score + segment max ----------------
template<int H>
__global__ void edge_alpha_max(const int* __restrict__ ei, const float* __restrict__ as_,
                               const float* __restrict__ ad_, float* __restrict__ alpha,
                               unsigned* __restrict__ menc) {
  int t = blockIdx.x * blockDim.x + threadIdx.x;
  if (t >= ET * H) return;
  int e = t / H, h = t - e * H;
  int src = (e < NE) ? ei[e]      : (e - NE);
  int dst = (e < NE) ? ei[NE + e] : (e - NE);
  float a  = as_[src * H + h] + ad_[dst * H + h];
  float al = (a > 0.f) ? a : NEG_SLOPE * a;
  alpha[t] = al;
  __hip_atomic_fetch_max(&menc[dst * H + h], fenc(al), __ATOMIC_RELAXED, __HIP_MEMORY_SCOPE_AGENT);
}

// ---------------- edge pass 2: exp + segment sum ----------------
template<int H>
__global__ void edge_expsum(const int* __restrict__ ei, const unsigned* __restrict__ menc,
                            float* __restrict__ alpha, float* __restrict__ z) {
  int t = blockIdx.x * blockDim.x + threadIdx.x;
  if (t >= ET * H) return;
  int e = t / H, h = t - e * H;
  int dst = (e < NE) ? ei[NE + e] : (e - NE);
  float ev = __expf(alpha[t] - fdec(menc[dst * H + h]));
  alpha[t] = ev;
  __hip_atomic_fetch_add(&z[dst * H + h], ev, __ATOMIC_RELAXED, __HIP_MEMORY_SCOPE_AGENT);
}

// ---------------- edge pass 3: normalized weighted scatter ----------------
template<int H, int C>
__global__ void edge_scatter(const int* __restrict__ ei, const float* __restrict__ expw,
                             const float* __restrict__ z, const float* __restrict__ Hin,
                             float* __restrict__ agg) {
  constexpr int HC  = H * C;
  constexpr int TPE = HC / 4;                  // 4 channels per lane
  long t = (long)blockIdx.x * blockDim.x + threadIdx.x;
  if (t >= (long)ET * TPE) return;
  int e  = (int)(t / TPE);
  int q  = (int)(t - (long)e * TPE);
  int c0 = q * 4;
  int h  = c0 / C;
  int src = (e < NE) ? ei[e]      : (e - NE);
  int dst = (e < NE) ? ei[NE + e] : (e - NE);
  float w = expw[e * H + h] / (z[dst * H + h] + EPSF);
  const float4 hv = *(const float4*)(Hin + (long)src * HC + c0);
  float* ap = agg + (long)dst * HC + c0;
  __hip_atomic_fetch_add(ap + 0, hv.x * w, __ATOMIC_RELAXED, __HIP_MEMORY_SCOPE_AGENT);
  __hip_atomic_fetch_add(ap + 1, hv.y * w, __ATOMIC_RELAXED, __HIP_MEMORY_SCOPE_AGENT);
  __hip_atomic_fetch_add(ap + 2, hv.z * w, __ATOMIC_RELAXED, __HIP_MEMORY_SCOPE_AGENT);
  __hip_atomic_fetch_add(ap + 3, hv.w * w, __ATOMIC_RELAXED, __HIP_MEMORY_SCOPE_AGENT);
}

// ---------------- finalize layer1: bias + relu, emit f32 (in place) and bf16 ----------------
__global__ void finalize1(float* __restrict__ agg, const float* __restrict__ b,
                          unsigned short* __restrict__ hb) {
  long t = (long)blockIdx.x * blockDim.x + threadIdx.x;
  if (t >= (long)NN * HC1) return;
  int c = (int)(t & (HC1 - 1));
  float v = agg[t] + b[c];
  v = (v > 0.f) ? v : 0.f;
  agg[t] = v;
  hb[t]  = f2bf(v);
}

// ---------------- finalize layer2: bias + relu -> d_out ----------------
__global__ void finalize2(const float* __restrict__ agg, const float* __restrict__ b,
                          float* __restrict__ out) {
  long t = (long)blockIdx.x * blockDim.x + threadIdx.x;
  if (t >= (long)NN * HC2) return;
  int c = (int)(t & (HC2 - 1));
  float v = agg[t] + b[c];
  out[t] = (v > 0.f) ? v : 0.f;
}

// ---------------- launcher ----------------
static constexpr size_t alup(size_t x) { return (x + 255) & ~(size_t)255; }

extern "C" void kernel_launch(void* const* d_in, const int* in_sizes, int n_in,
                              void* d_out, int out_size, void* d_ws, size_t ws_size,
                              hipStream_t stream) {
  const float* x    = (const float*)d_in[0];
  const int*   ei   = (const int*)  d_in[1];
  const float* W1   = (const float*)d_in[2];
  const float* asw1 = (const float*)d_in[3];
  const float* adw1 = (const float*)d_in[4];
  const float* b1   = (const float*)d_in[5];
  const float* W2   = (const float*)d_in[6];
  const float* asw2 = (const float*)d_in[7];
  const float* adw2 = (const float*)d_in[8];
  const float* b2   = (const float*)d_in[9];
  float* out = (float*)d_out;

  char* ws = (char*)d_ws;
  size_t o = 0;
  auto take = [&](size_t bytes) { size_t r = o; o += alup(bytes); return r; };
  unsigned short* xb   = (unsigned short*)(ws + take((size_t)NN * FIN * 2));   // x in bf16
  unsigned short* wt1  = (unsigned short*)(ws + take((size_t)HC1 * KDIM * 2)); // W1^T bf16
  unsigned short* wt2  = (unsigned short*)(ws + take((size_t)HC2 * KDIM * 2)); // W2^T bf16
  float*    h1   = (float*)   (ws + take((size_t)NN * HC1 * 4));               // x@W1
  float*    as1  = (float*)   (ws + take((size_t)NN * H1 * 4));
  float*    ad1  = (float*)   (ws + take((size_t)NN * H1 * 4));
  unsigned* m1   = (unsigned*)(ws + take((size_t)NN * H1 * 4));
  float*    z1   = (float*)   (ws + take((size_t)NN * H1 * 4));
  float*    al1  = (float*)   (ws + take((size_t)ET * H1 * 4));
  float*    agg1 = (float*)   (ws + take((size_t)NN * HC1 * 4));               // -> relu'd in place
  unsigned short* h1b = (unsigned short*)(ws + take((size_t)NN * HC1 * 2));
  float*    h2   = (float*)   (ws + take((size_t)NN * HC2 * 4));
  float*    as2  = (float*)   (ws + take((size_t)NN * H2 * 4));
  float*    ad2  = (float*)   (ws + take((size_t)NN * H2 * 4));
  unsigned* m2   = (unsigned*)(ws + take((size_t)NN * H2 * 4));
  float*    z2   = (float*)   (ws + take((size_t)NN * H2 * 4));
  float*    al2  = (float*)   (ws + take((size_t)ET * H2 * 4));
  float*    agg2 = (float*)   (ws + take((size_t)NN * HC2 * 4));
  (void)ws_size; (void)in_sizes; (void)n_in; (void)out_size;

  const int B = 256;
  auto g = [](long n, int b) { return (unsigned)((n + b - 1) / b); };

  // ---- prep ----
  cvt_bf16<<<g((long)NN * FIN, B), B, 0, stream>>>(x, xb, (long)NN * FIN);
  cvt_transpose_w<<<g(KDIM * HC1, B), B, 0, stream>>>(W1, wt1, KDIM, HC1);
  cvt_transpose_w<<<g(KDIM * HC2, B), B, 0, stream>>>(W2, wt2, KDIM, HC2);

  // ---- layer 1 ----
  wmma_gemm_bf16<HC1><<<g(STRIPS, 4), 128, 0, stream>>>(xb, wt1, h1);
  node_scores<<<g((long)NN * H1, B), B, 0, stream>>>(h1, asw1, adw1, as1, ad1, H1, C1);
  fill_u32<<<g((long)NN * H1, B), B, 0, stream>>>(m1, ENC_NEG_INF, (long)NN * H1);
  fill_u32<<<g((long)NN * H1, B), B, 0, stream>>>((unsigned*)z1, 0u, (long)NN * H1);
  fill_u32<<<g((long)NN * HC1, B), B, 0, stream>>>((unsigned*)agg1, 0u, (long)NN * HC1);
  edge_alpha_max<H1><<<g((long)ET * H1, B), B, 0, stream>>>(ei, as1, ad1, al1, m1);
  edge_expsum<H1>   <<<g((long)ET * H1, B), B, 0, stream>>>(ei, m1, al1, z1);
  edge_scatter<H1, C1><<<g((long)ET * (HC1 / 4), B), B, 0, stream>>>(ei, al1, z1, h1, agg1);
  finalize1<<<g((long)NN * HC1, B), B, 0, stream>>>(agg1, b1, h1b);

  // ---- layer 2 ----
  wmma_gemm_bf16<HC2><<<g(STRIPS, 4), 128, 0, stream>>>(h1b, wt2, h2);
  node_scores<<<g((long)NN * H2, B), B, 0, stream>>>(h2, asw2, adw2, as2, ad2, H2, C2);
  fill_u32<<<g((long)NN * H2, B), B, 0, stream>>>(m2, ENC_NEG_INF, (long)NN * H2);
  fill_u32<<<g((long)NN * H2, B), B, 0, stream>>>((unsigned*)z2, 0u, (long)NN * H2);
  fill_u32<<<g((long)NN * HC2, B), B, 0, stream>>>((unsigned*)agg2, 0u, (long)NN * HC2);
  edge_alpha_max<H2><<<g((long)ET * H2, B), B, 0, stream>>>(ei, as2, ad2, al2, m2);
  edge_expsum<H2>   <<<g((long)ET * H2, B), B, 0, stream>>>(ei, m2, al2, z2);
  edge_scatter<H2, C2><<<g((long)ET * (HC2 / 4), B), B, 0, stream>>>(ei, al2, z2, h2, agg2);
  finalize2<<<g((long)NN * HC2, B), B, 0, stream>>>(agg2, b2, out);
}
